// MultiScaleRetention_67001489817984
// MI455X (gfx1250) — compile-verified, matrix operand order checked by reading, not run on hardware
//
#include <hip/hip_runtime.h>
#include <math.h>

typedef __bf16 bf16;
typedef __attribute__((ext_vector_type(16))) __bf16 v16bf;
typedef __attribute__((ext_vector_type(8)))  __bf16 v8bf;
typedef __attribute__((ext_vector_type(8)))  float  v8f;
typedef __attribute__((ext_vector_type(4)))  unsigned int v4u;
typedef __attribute__((ext_vector_type(4)))  unsigned int u32x4;
typedef __attribute__((ext_vector_type(8)))  int i32x8;
typedef __attribute__((ext_vector_type(4)))  int i32x4;

#define NB  2
#define NT  2048
#define NE  1024
#define NH  8
#define NDK 128
#define NDV 2048
#define NDH 256
#define BT  (NB*NT)   // 4096

#if __has_builtin(__builtin_amdgcn_tensor_load_to_lds) && __has_builtin(__builtin_amdgcn_s_wait_tensorcnt)
#define HAS_TDM 1
#else
#define HAS_TDM 0
#endif

// ---------------------------------------------------------------------------
// WMMA helpers (CDNA5 16x16x32 bf16, f32 accumulate)
// ---------------------------------------------------------------------------
__device__ inline v8f wmma_bf16(v16bf a, v16bf b, v8f c) {
  return __builtin_amdgcn_wmma_f32_16x16x32_bf16(false, a, false, b, (short)0, c, false, false);
}

// 16x32 bf16 A/B fragment row from LDS (ISA 7.12.2 layout):
// lane (hi=lane/16): element j -> k = kbase + (j<8 ? 8*hi+j : 16+8*hi+(j-8))
__device__ inline v16bf load_frag(const bf16* base, int row, int stride, int kbase, int hi) {
  const bf16* p = base + row * stride + kbase + 8 * hi;
  v16bf r;
  ((v8bf*)&r)[0] = *(const v8bf*)p;
  ((v8bf*)&r)[1] = *(const v8bf*)(p + 16);
  return r;
}

// ---------------------------------------------------------------------------
// Tensor Data Mover: 2D tile global->LDS with hardware padding.
// Builds the D# per CDNA5 ISA; issued per-wave,
// tracked by TENSORcnt. padi/pada are the encoded pad_interval/pad_amount.
// ---------------------------------------------------------------------------
#if HAS_TDM
__device__ inline unsigned lds_off(const void* p) {
  return (unsigned)(unsigned long long)(__attribute__((address_space(3))) const char*)p;
}
__device__ inline void tdm_load_2d(unsigned ldsoff, const void* gp,
                                   unsigned width, unsigned rows, unsigned stride,
                                   unsigned padi, unsigned pada)
{
  unsigned long long a = (unsigned long long)gp;
  u32x4 g0;
  g0[0] = 1u;                                                // count=1, user desc
  g0[1] = ldsoff;                                            // lds_addr (bytes)
  g0[2] = (unsigned)a;                                       // global_addr[31:0]
  g0[3] = ((unsigned)(a >> 32) & 0x1FFFFFFu) | (2u << 30);   // addr[56:32] | type=2
  i32x8 g1;
  g1[0] = (int)((1u << 16) | (1u << 20) | (padi << 22) | (pada << 25)); // 2B, pad_en
  g1[1] = (int)((width & 0xffffu) << 16);                    // tensor_dim0[15:0]
  g1[2] = (int)((width >> 16) | ((rows & 0xffffu) << 16));   // dim0 hi | tensor_dim1 lo
  g1[3] = (int)((rows >> 16) | ((width & 0xffffu) << 16));   // dim1 hi | tile_dim0
  g1[4] = (int)(rows & 0xffffu);                             // tile_dim1 (tile_dim2=0)
  g1[5] = (int)stride;                                       // tensor_dim0_stride lo32
  g1[6] = 0;
  g1[7] = 0;
  i32x4 z4 = (i32x4){};
#if __clang_major__ >= 23
  i32x8 z8 = (i32x8){};
  __builtin_amdgcn_tensor_load_to_lds(g0, g1, z4, z4, z8, 0);
#else
  __builtin_amdgcn_tensor_load_to_lds(g0, g1, z4, z4, 0);
#endif
}
#endif

// ---------------------------------------------------------------------------
// f32 -> bf16 conversion
// ---------------------------------------------------------------------------
__global__ void cvt_kernel(const float* __restrict__ src, bf16* __restrict__ dst, int n) {
  int i = blockIdx.x * 256 + threadIdx.x;
  if (i < n) dst[i] = (bf16)src[i];
}

// ---------------------------------------------------------------------------
// Generic bf16 GEMM: C[m,n] = sum_k A[m,k] * W[n,k]
// A: (M,K) row-major bf16, W: (N,K) row-major bf16.
// Output: bf16 if obf!=0 else f32 (row-major M x N).
// Block tile 128x128, k-step 32, 8 waves. TDM path: double-buffered LDS,
// wave 0 issues next tiles while all waves run WMMA on current buffer.
// ---------------------------------------------------------------------------
#define GLDT 40   // padded LDS row stride (bf16): 16 DW row + 4 DW pad

__global__ __launch_bounds__(256) void gemm_bf16_kernel(
    const bf16* __restrict__ A, const bf16* __restrict__ W,
    void* __restrict__ Cout, int M, int N, int K, int obf)
{
  int tid  = threadIdx.x;
  int lane = tid & 31, wid = tid >> 5;
  int hi = lane >> 4, ln = lane & 15;
  int bm = blockIdx.x * 128;
  int bn = blockIdx.y * 128;
  int wm = (wid & 3) * 32;
  int wn = (wid >> 2) * 64;

  v8f acc[2][4];
  for (int i = 0; i < 2; i++)
    for (int j = 0; j < 4; j++) acc[i][j] = (v8f){};

#if HAS_TDM
  __shared__ bf16 sA[2][128 * GLDT];
  __shared__ bf16 sB[2][128 * GLDT];
  int nsteps = K >> 5;
  if (wid == 0) {
    tdm_load_2d(lds_off(&sA[0][0]), A + (size_t)bm * K, 32, 128, (unsigned)K, 3, 3);
    tdm_load_2d(lds_off(&sB[0][0]), W + (size_t)bn * K, 32, 128, (unsigned)K, 3, 3);
  }
  for (int i = 0; i < nsteps; i++) {
    int cur = i & 1;
    if (wid == 0) {
      if (i + 1 < nsteps) {
        int k1 = (i + 1) << 5;
        tdm_load_2d(lds_off(&sA[cur ^ 1][0]), A + (size_t)bm * K + k1, 32, 128, (unsigned)K, 3, 3);
        tdm_load_2d(lds_off(&sB[cur ^ 1][0]), W + (size_t)bn * K + k1, 32, 128, (unsigned)K, 3, 3);
        __builtin_amdgcn_s_wait_tensorcnt(2);   // retire current buffer only
      } else {
        __builtin_amdgcn_s_wait_tensorcnt(0);
      }
    }
    __syncthreads();
    v16bf af[2], bf_[4];
#pragma unroll
    for (int x = 0; x < 2; x++) af[x]  = load_frag(sA[cur], wm + 16 * x + ln, GLDT, 0, hi);
#pragma unroll
    for (int j = 0; j < 4; j++) bf_[j] = load_frag(sB[cur], wn + 16 * j + ln, GLDT, 0, hi);
#pragma unroll
    for (int x = 0; x < 2; x++)
#pragma unroll
      for (int j = 0; j < 4; j++)
        acc[x][j] = wmma_bf16(af[x], bf_[j], acc[x][j]);
    __syncthreads();
  }
#else
  __shared__ bf16 sA[128 * GLDT];
  __shared__ bf16 sB[128 * GLDT];
  int lr = tid >> 1;
  int lc = (tid & 1) * 16;
  for (int k0 = 0; k0 < K; k0 += 32) {
    __syncthreads();
    const bf16* ga = A + (size_t)(bm + lr) * K + k0 + lc;
    *(v4u*)&sA[lr * GLDT + lc]     = *(const v4u*)ga;
    *(v4u*)&sA[lr * GLDT + lc + 8] = *(const v4u*)(ga + 8);
    const bf16* gw = W + (size_t)(bn + lr) * K + k0 + lc;
    *(v4u*)&sB[lr * GLDT + lc]     = *(const v4u*)gw;
    *(v4u*)&sB[lr * GLDT + lc + 8] = *(const v4u*)(gw + 8);
    __syncthreads();
    v16bf af[2], bf_[4];
#pragma unroll
    for (int x = 0; x < 2; x++) af[x]  = load_frag(sA, wm + 16 * x + ln, GLDT, 0, hi);
#pragma unroll
    for (int j = 0; j < 4; j++) bf_[j] = load_frag(sB, wn + 16 * j + ln, GLDT, 0, hi);
#pragma unroll
    for (int x = 0; x < 2; x++)
#pragma unroll
      for (int j = 0; j < 4; j++)
        acc[x][j] = wmma_bf16(af[x], bf_[j], acc[x][j]);
  }
#endif

  // C fragment: VGPR r, lane -> (m = 16*i + 8*hi + r, n = 16*j + ln)
#pragma unroll
  for (int x = 0; x < 2; x++)
#pragma unroll
    for (int j = 0; j < 4; j++) {
      int mb = bm + wm + 16 * x + 8 * hi;
      int nb = bn + wn + 16 * j + ln;
      if (obf) {
        bf16* Cb = (bf16*)Cout;
#pragma unroll
        for (int r = 0; r < 8; r++)
          Cb[(size_t)(mb + r) * N + nb] = (bf16)acc[x][j][r];
      } else {
        float* Cf = (float*)Cout;
#pragma unroll
        for (int r = 0; r < 8; r++)
          Cf[(size_t)(mb + r) * N + nb] = acc[x][j][r];
      }
    }
}

// ---------------------------------------------------------------------------
// RoPE (theta_shift) + head split: x bf16 (b,t,h*DK+d) -> bf16 (b,h,t,d)
// rotate_every_two: even d -> -x[d+1], odd d -> x[d-1]
// ---------------------------------------------------------------------------
__global__ void rope_kernel(const bf16* __restrict__ x,
                            const float* __restrict__ sinp,
                            const float* __restrict__ cosp,
                            bf16* __restrict__ outp, float scale)
{
  int idx = blockIdx.x * 256 + threadIdx.x;   // over NB*NT*NE
  if (idx >= NB * NT * NE) return;
  int d = idx & (NDK - 1);
  int h = (idx >> 7)  & (NH - 1);
  int t = (idx >> 10) & (NT - 1);
  int b = idx >> 21;
  float v  = (float)x[idx];
  float vp = (float)x[idx ^ 1];
  float rot = (d & 1) ? vp : -vp;
  float r = (v * cosp[t * NDK + d] + rot * sinp[t * NDK + d]) * scale;
  outp[(((size_t)(b * NH + h) * NT) + t) * NDK + d] = (bf16)r;
}

// v bf16 (b,t,h*DH+d) -> bf16 (b,h,t,d)
__global__ void vperm_kernel(const bf16* __restrict__ v, bf16* __restrict__ outp) {
  int idx = blockIdx.x * 256 + threadIdx.x;   // over NB*NT*NDV
  if (idx >= NB * NT * NDV) return;
  int d = idx & (NDH - 1);
  int h = (idx >> 8)  & (NH - 1);
  int t = (idx >> 11) & (NT - 1);
  int b = idx >> 22;
  outp[(((size_t)(b * NH + h) * NT) + t) * NDH + d] = v[idx];
}

// ---------------------------------------------------------------------------
// Fused retention per (b,h) and 64-row t-block:
//   pass 1: rowsum_t = sum_s (qr_t . kr_s) * mask[h,t,s]   (causal cutoff)
//   denom  = clamp(|rowsum|, 1, 5e4)
//   pass 2: out = ((qr krT) * mask / denom) @ v   (WMMA chain via LDS relayout)
//   epilogue: RMS-norm over DH, silu(g) gating, emit gated bf16 (b,t,DV)
// ---------------------------------------------------------------------------
__global__ __launch_bounds__(256) void retention_kernel(
    const bf16* __restrict__ qr, const bf16* __restrict__ kr,
    const bf16* __restrict__ vb, const bf16* __restrict__ g,
    const float* __restrict__ mask, bf16* __restrict__ gated)
{
  __shared__ bf16 sQ[64 * 136];    // qr block rows (64 x 128), stride 136
  __shared__ bf16 sK[32 * 136];    // kr s-chunk   (32 x 128), stride 136
  __shared__ bf16 sV[256 * 40];    // v chunk transposed (d, s) (256 x 32)
  __shared__ bf16 sR[8 * 16 * 40]; // per-wave retention subtile relayout
  __shared__ float rowsum[64];
  __shared__ float sumsq[64];
  __shared__ float denom[64];

  int tid = threadIdx.x, lane = tid & 31, wid = tid >> 5;
  int hi = lane >> 4, ln = lane & 15;
  int tb = blockIdx.x, bh = blockIdx.y;
  int b = bh >> 3, h = bh & 7;
  int t0 = tb * 64;

  const bf16* qr_bh = qr + (size_t)bh * NT * NDK;
  const bf16* kr_bh = kr + (size_t)bh * NT * NDK;
  const bf16* vb_bh = vb + (size_t)bh * NT * NDH;
  const float* mask_h = mask + (size_t)h * NT * NT;

  // stage qr rows t0..t0+63
#if HAS_TDM
  if (wid == 0)
    tdm_load_2d(lds_off(sQ), qr_bh + (size_t)t0 * NDK, 128, 64, 128, 5, 3);
  if (tid < 64) { rowsum[tid] = 0.f; sumsq[tid] = 0.f; }
  if (wid == 0) __builtin_amdgcn_s_wait_tensorcnt(0);
#else
  {
    int r = tid >> 2;
    int c = (tid & 3) * 32;
    const bf16* p = qr_bh + (size_t)(t0 + r) * NDK + c;
#pragma unroll
    for (int q = 0; q < 4; q++)
      *(v4u*)&sQ[r * 136 + c + 8 * q] = *(const v4u*)(p + 8 * q);
  }
  if (tid < 64) { rowsum[tid] = 0.f; sumsq[tid] = 0.f; }
#endif
  __syncthreads();

  int mt = wid & 3;       // row tile (16 rows each)
  int ch = wid >> 2;      // pass1: s-subtile; pass2: d-half
  int s_end = t0 + 64;    // causal cutoff: mask==0 for s > t

  // hoist A fragments (qr rows, K=128 -> 4 frags); sQ never overwritten
  v16bf qfrag[4];
#pragma unroll
  for (int kk = 0; kk < 4; kk++)
    qfrag[kk] = load_frag(sQ, 16 * mt + ln, 136, 32 * kk, hi);

  // ---------------- pass 1: masked row sums ----------------
  float psum[8];
#pragma unroll
  for (int r = 0; r < 8; r++) psum[r] = 0.f;

  for (int s0 = 0; s0 < s_end; s0 += 32) {
#if HAS_TDM
    if (wid == 0) {
      tdm_load_2d(lds_off(sK), kr_bh + (size_t)s0 * NDK, 128, 32, 128, 5, 3);
      __builtin_amdgcn_s_wait_tensorcnt(0);
    }
#else
    {
      int r = tid >> 3;
      int c = (tid & 7) * 16;
      const bf16* p = kr_bh + (size_t)(s0 + r) * NDK + c;
      *(v4u*)&sK[r * 136 + c]     = *(const v4u*)p;
      *(v4u*)&sK[r * 136 + c + 8] = *(const v4u*)(p + 8);
    }
#endif
    __syncthreads();
    v8f cacc = (v8f){};
#pragma unroll
    for (int kk = 0; kk < 4; kk++) {
      v16bf bb = load_frag(sK, 16 * ch + ln, 136, 32 * kk, hi);
      cacc = wmma_bf16(qfrag[kk], bb, cacc);
    }
#pragma unroll
    for (int r = 0; r < 8; r++) {
      int t = t0 + 16 * mt + 8 * hi + r;
      int s = s0 + 16 * ch + ln;
      psum[r] += cacc[r] * mask_h[(size_t)t * NT + s];
    }
    __syncthreads();
  }
#pragma unroll
  for (int r = 0; r < 8; r++) {
    float v = psum[r];
    for (int off = 1; off < 16; off <<= 1) v += __shfl_xor(v, off, 32);
    if (ln == 0) atomicAdd(&rowsum[16 * mt + 8 * hi + r], v);
  }
  __syncthreads();
  if (tid < 64) denom[tid] = fminf(fmaxf(fabsf(rowsum[tid]), 1.0f), 50000.0f);
  __syncthreads();

  // ---------------- pass 2: normalized retention @ v ----------------
  v8f oacc[8];
#pragma unroll
  for (int j = 0; j < 8; j++) oacc[j] = (v8f){};
  bf16* sRw = &sR[wid * 16 * 40];

  for (int s0 = 0; s0 < s_end; s0 += 32) {
#if HAS_TDM
    if (wid == 0)
      tdm_load_2d(lds_off(sK), kr_bh + (size_t)s0 * NDK, 128, 32, 128, 5, 3);
#else
    {
      int r = tid >> 3;
      int c = (tid & 7) * 16;
      const bf16* p = kr_bh + (size_t)(s0 + r) * NDK + c;
      *(v4u*)&sK[r * 136 + c]     = *(const v4u*)p;
      *(v4u*)&sK[r * 136 + c + 8] = *(const v4u*)(p + 8);
    }
#endif
    {
      // transpose-stage v chunk: sV[d][i] = v[s0+i, d]
      int i = tid >> 3;
      int c = (tid & 7) * 32;
      const bf16* p = vb_bh + (size_t)(s0 + i) * NDH + c;
      bf16 tmp[32];
#pragma unroll
      for (int q = 0; q < 4; q++) *(v4u*)&tmp[8 * q] = *(const v4u*)(p + 8 * q);
#pragma unroll
      for (int q = 0; q < 32; q++) sV[(c + q) * 40 + i] = tmp[q];
    }
#if HAS_TDM
    if (wid == 0) __builtin_amdgcn_s_wait_tensorcnt(0);
#endif
    __syncthreads();

    // retention subtile 16x32 (both d-half waves recompute it)
    v8f rc[2];
#pragma unroll
    for (int n = 0; n < 2; n++) {
      v8f cacc = (v8f){};
#pragma unroll
      for (int kk = 0; kk < 4; kk++) {
        v16bf bb = load_frag(sK, 16 * n + ln, 136, 32 * kk, hi);
        cacc = wmma_bf16(qfrag[kk], bb, cacc);
      }
      rc[n] = cacc;
    }
    // scale by mask/denom, write to per-wave LDS (C-frag -> A-frag relayout)
#pragma unroll
    for (int n = 0; n < 2; n++)
#pragma unroll
      for (int r = 0; r < 8; r++) {
        int rl = 8 * hi + r;
        int t  = t0 + 16 * mt + rl;
        int s  = s0 + 16 * n + ln;
        float val = rc[n][r] * mask_h[(size_t)t * NT + s] / denom[16 * mt + rl];
        sRw[rl * 40 + 16 * n + ln] = (bf16)val;
      }
    __builtin_amdgcn_wave_barrier();
    asm volatile("s_wait_dscnt 0" ::: "memory");   // same-wave LDS RAW fence

    v16bf afr = load_frag(sRw, ln, 40, 0, hi);     // A frag: 16 rows x K=32(s)
#pragma unroll
    for (int j = 0; j < 8; j++) {                  // d tiles in this wave's half
      v16bf bb = load_frag(sV, 128 * ch + 16 * j + ln, 40, 0, hi);
      oacc[j] = wmma_bf16(afr, bb, oacc[j]);
    }
    __syncthreads();
  }

  // ---------------- epilogue: RMS norm + silu(g) gating ----------------
  float psq[8];
#pragma unroll
  for (int r = 0; r < 8; r++) psq[r] = 0.f;
#pragma unroll
  for (int j = 0; j < 8; j++)
#pragma unroll
    for (int r = 0; r < 8; r++) psq[r] += oacc[j][r] * oacc[j][r];
#pragma unroll
  for (int r = 0; r < 8; r++) {
    float v = psq[r];
    for (int off = 1; off < 16; off <<= 1) v += __shfl_xor(v, off, 32);
    if (ln == 0) atomicAdd(&sumsq[16 * mt + 8 * hi + r], v);
  }
  __syncthreads();

#pragma unroll
  for (int j = 0; j < 8; j++)
#pragma unroll
    for (int r = 0; r < 8; r++) {
      int rl = 8 * hi + r;
      int t  = t0 + 16 * mt + rl;
      int d  = 128 * ch + 16 * j + ln;
      float rms = rsqrtf(sumsq[16 * mt + rl] * (1.0f / (float)NDH) + 1e-6f);
      float o = oacc[j][r] * rms;
      size_t gi = ((size_t)(b * NT + t)) * NDV + h * NDH + d;
      float gv = (float)g[gi];
      float sg = gv / (1.f + __expf(-gv));
      gated[gi] = (bf16)(sg * o);
    }
}

// ---------------------------------------------------------------------------
// Host launch
// ---------------------------------------------------------------------------
extern "C" void kernel_launch(void* const* d_in, const int* in_sizes, int n_in,
                              void* d_out, int out_size, void* d_ws, size_t ws_size,
                              hipStream_t stream)
{
  const float* hs   = (const float*)d_in[0];
  const float* sinp = (const float*)d_in[1];
  const float* cosp = (const float*)d_in[2];
  const float* mask = (const float*)d_in[3];
  const float* Wq   = (const float*)d_in[4];
  const float* Wk   = (const float*)d_in[5];
  const float* Wv   = (const float*)d_in[6];
  const float* Wg   = (const float*)d_in[7];
  const float* Wo   = (const float*)d_in[8];
  float* outp = (float*)d_out;

  char* ws = (char*)d_ws;
  size_t off = 0;
  auto alloc = [&](size_t bytes) -> void* {
    void* p = ws + off;
    off += (bytes + 255) & ~(size_t)255;
    return p;
  };
  bf16* hsb = (bf16*)alloc((size_t)BT * NE * 2);
  bf16* wqb = (bf16*)alloc((size_t)NE * NE * 2);
  bf16* wkb = (bf16*)alloc((size_t)NE * NE * 2);
  bf16* wvb = (bf16*)alloc((size_t)NDV * NE * 2);
  bf16* wgb = (bf16*)alloc((size_t)NDV * NE * 2);
  bf16* wob = (bf16*)alloc((size_t)NE * NDV * 2);
  bf16* qb  = (bf16*)alloc((size_t)BT * NE * 2);
  bf16* kb  = (bf16*)alloc((size_t)BT * NE * 2);
  bf16* vbm = (bf16*)alloc((size_t)BT * NDV * 2);
  bf16* gb  = (bf16*)alloc((size_t)BT * NDV * 2);
  bf16* qrb = (bf16*)alloc((size_t)BT * NE * 2);
  bf16* krb = (bf16*)alloc((size_t)BT * NE * 2);
  bf16* vbb = (bf16*)alloc((size_t)BT * NDV * 2);
  bf16* gtd = (bf16*)alloc((size_t)BT * NDV * 2);

  auto cvt = [&](const float* s, bf16* d, int n) {
    cvt_kernel<<<(n + 255) / 256, 256, 0, stream>>>(s, d, n);
  };
  cvt(hs, hsb, BT * NE);
  cvt(Wq, wqb, NE * NE);
  cvt(Wk, wkb, NE * NE);
  cvt(Wv, wvb, NDV * NE);
  cvt(Wg, wgb, NDV * NE);
  cvt(Wo, wob, NE * NDV);

  // projections: bf16 outputs
  gemm_bf16_kernel<<<dim3(BT / 128, NE / 128), 256, 0, stream>>>(hsb, wqb, qb, BT, NE, NE, 1);
  gemm_bf16_kernel<<<dim3(BT / 128, NE / 128), 256, 0, stream>>>(hsb, wkb, kb, BT, NE, NE, 1);
  gemm_bf16_kernel<<<dim3(BT / 128, NDV / 128), 256, 0, stream>>>(hsb, wvb, vbm, BT, NDV, NE, 1);
  gemm_bf16_kernel<<<dim3(BT / 128, NDV / 128), 256, 0, stream>>>(hsb, wgb, gb, BT, NDV, NE, 1);

  // RoPE + head split (k carries DK^-0.5 scaling), v permute
  int nqe = NB * NT * NE;
  rope_kernel<<<(nqe + 255) / 256, 256, 0, stream>>>(qb, sinp, cosp, qrb, 1.0f);
  rope_kernel<<<(nqe + 255) / 256, 256, 0, stream>>>(kb, sinp, cosp, krb, 0.08838834764831845f);
  int nve = NB * NT * NDV;
  vperm_kernel<<<(nve + 255) / 256, 256, 0, stream>>>(vbm, vbb);

  // fused retention + RMS norm + silu gating
  retention_kernel<<<dim3(NT / 64, NB * NH), 256, 0, stream>>>(qrb, krb, vbb, gb, mask, gtd);

  // output projection (f32 to d_out): out = gated @ Wo^T
  gemm_bf16_kernel<<<dim3(BT / 128, NE / 128), 256, 0, stream>>>(gtd, wob, outp, BT, NE, NDV, 0);
}
